// RNNPPO_11802570129991
// MI455X (gfx1250) — compile-verified, matrix operand order checked by reading, not run on hardware
//
#include <hip/hip_runtime.h>
#include <hip/hip_bf16.h>
#include <math.h>

#define T_    512
#define B_    128
#define D_    256
#define H_    512
#define MID_  512
#define G4H   (4 * H_)      // 2048
#define KCAT  (D_ + H_)     // 768

#define WSTRIDE 776   // u16 stride for weight LDS rows (16B aligned, bank-staggered)
#define ASTRIDE 40    // u16 stride for A-chunk LDS rows (64B data + 16B pad)
#define GSTRIDE 132   // f32 stride for gate LDS rows
#define YSTRIDE 520   // u16 stride for MLP activation LDS rows
#define LSTM_BLOCKS 16

typedef __attribute__((ext_vector_type(16))) __bf16 v16bf;
typedef __attribute__((ext_vector_type(8)))  float  v8f;
typedef unsigned int  v4u __attribute__((ext_vector_type(4)));
typedef int           v4i __attribute__((ext_vector_type(4)));
typedef int           v8i __attribute__((ext_vector_type(8)));
typedef unsigned short u16;
typedef unsigned int   u32;

__device__ __forceinline__ u16 f2bf(float f) {
  u32 u = __float_as_uint(f);
  u32 r = u + 0x7FFFu + ((u >> 16) & 1u);   // round-to-nearest-even
  return (u16)(r >> 16);
}
__device__ __forceinline__ float sigm(float x) { return 1.0f / (1.0f + __expf(-x)); }

__device__ __forceinline__ v8f wmma_bf16(v16bf a, v16bf b, v8f c) {
  return __builtin_amdgcn_wmma_f32_16x16x32_bf16(false, a, false, b, (short)0, c,
                                                 false, false);
}

// ---------------------------------------------------------------------------
// Tensor Data Mover: load a [tile_h rows x 32 bf16] 2D tile (global row stride
// `gstride` elements) into LDS at `lds_off`, padding each 16-DWORD row with
// 4 DWORDs so rows land at the ASTRIDE(=40 u16 = 20 DWORD) LDS pitch.
// Must be executed by a single wave (EXEC ignored; wave-level DMA).
// ---------------------------------------------------------------------------
__device__ __forceinline__ void tdm_load_tile32_bf16(const u16* gsrc, u32 lds_off,
                                                     int tile_h, int gstride) {
  unsigned long long ga = (unsigned long long)(size_t)gsrc;
  v4u g0;
  g0.x = 1u;                                        // count=1 (valid user D#)
  g0.y = lds_off;                                   // LDS byte address
  g0.z = (u32)(ga & 0xFFFFFFFFu);                   // global_addr[31:0]
  g0.w = (u32)((ga >> 32) & 0x01FFFFFFu) | (2u << 30);  // addr[56:32] | type=2
  v8i g1;
  u32 w0 = (1u << 16)    // data_size = 2 bytes
         | (1u << 20)    // pad_enable
         | (3u << 22)    // pad_interval: 16 DWORDs of data...
         | (3u << 25);   // ...then 4 DWORDs of padding
  g1[0] = (int)w0;
  g1[1] = (int)((32u & 0xFFFFu) << 16);                         // tensor_dim0 lo16 = 32
  g1[2] = (int)(((32u >> 16) & 0xFFFFu) |
                (((u32)tile_h & 0xFFFFu) << 16));               // dim0 hi | dim1 lo
  g1[3] = (int)(((((u32)tile_h) >> 16) & 0xFFFFu) |
                (32u << 16));                                   // dim1 hi | tile_dim0=32
  g1[4] = (int)((u32)tile_h & 0xFFFFu);                         // tile_dim1 (tile_dim2=0)
  g1[5] = (int)(u32)gstride;                                    // tensor_dim0_stride lo32
  g1[6] = 0;
  g1[7] = 0;
  v4i gz4 = (v4i){0, 0, 0, 0};
  v8i gz8 = (v8i){0, 0, 0, 0, 0, 0, 0, 0};
  __builtin_amdgcn_tensor_load_to_lds(g0, g1, gz4, gz4, gz8, 0);
}

// ---------------------------------------------------------------------------
// Prep: pack [W_ih | W_hh] -> bf16, bias sum, h0 -> bf16, zero grid-barrier ctr
// ---------------------------------------------------------------------------
__global__ void k_prep_lstm(const float* __restrict__ W_ih,
                            const float* __restrict__ W_hh,
                            const float* __restrict__ b_ih,
                            const float* __restrict__ b_hh,
                            const float* __restrict__ hidden,
                            u16* __restrict__ wcat, float* __restrict__ bsum,
                            u16* __restrict__ hbuf, u32* __restrict__ cnt) {
  size_t i = (size_t)blockIdx.x * blockDim.x + threadIdx.x;
  size_t total = (size_t)G4H * KCAT;
  size_t stride = (size_t)gridDim.x * blockDim.x;
  for (size_t idx = i; idx < total; idx += stride) {
    int r = (int)(idx / KCAT), c = (int)(idx % KCAT);
    float v = (c < D_) ? W_ih[(size_t)r * D_ + c] : W_hh[(size_t)r * H_ + (c - D_)];
    wcat[idx] = f2bf(v);
  }
  if (i < (size_t)G4H) bsum[i] = b_ih[i] + b_hh[i];
  if (i < (size_t)B_ * H_) hbuf[i] = f2bf(hidden[i]);
  if (i == 0) *cnt = 0u;
}

__global__ void k_cvt(u16* __restrict__ dst, const float* __restrict__ src, int n) {
  int i = blockIdx.x * blockDim.x + threadIdx.x;
  for (; i < n; i += gridDim.x * blockDim.x) dst[i] = f2bf(src[i]);
}

// zero-pad small head weight (rows x 512) into 16 x 512
__global__ void k_pad16(u16* __restrict__ dst, const float* __restrict__ src, int rows) {
  int i = blockIdx.x * blockDim.x + threadIdx.x;
  if (i < 16 * MID_) {
    int r = i / MID_;
    dst[i] = (r < rows) ? f2bf(src[i]) : (u16)0;
  }
}

// ---------------------------------------------------------------------------
// Ragged packing: prefix sums + row map  n -> b*T + t
// ---------------------------------------------------------------------------
__global__ void k_offsets(const int* __restrict__ len, int* __restrict__ offs) {
  if (blockIdx.x == 0 && threadIdx.x == 0) {
    int run = 0;
    for (int b = 0; b < B_; ++b) {
      offs[b] = run;
      int l = len[b];
      if (l < 0) l = 0;
      if (l > T_) l = T_;
      run += l;
    }
    offs[B_] = run;
  }
}

__global__ void k_rowmap(const int* __restrict__ len, const int* __restrict__ offs,
                         int* __restrict__ rowmap) {
  int b = blockIdx.x;
  int l = len[b];
  if (l < 0) l = 0;
  if (l > T_) l = T_;
  int o = offs[b];
  for (int i = threadIdx.x; i < l; i += blockDim.x) rowmap[o + i] = b * T_ + i;
}

// ---------------------------------------------------------------------------
// Persistent LSTM: 16 blocks, weights LDS-resident, grid barrier per step.
// Block gb owns h columns [gb*32, gb*32+32) i.e. gate rows {q*512+gb*32+j}.
// h_prev chunks are DMA'd into LDS by the Tensor Data Mover.
// ---------------------------------------------------------------------------
__global__ void __launch_bounds__(256)
k_lstm(const float* __restrict__ state, const float* __restrict__ cell,
       const u16* __restrict__ wcat, const float* __restrict__ bsum,
       u16* __restrict__ hbuf,   // ping-pong [2][B_][H_] bf16
       u16* __restrict__ hs,     // [T_][B_][H_] bf16
       u32* __restrict__ cnt) {
  extern __shared__ char smem[];
  u16*   Wlds = (u16*)smem;                                            // [128][WSTRIDE]
  u16*   Alds = (u16*)(smem + 128 * WSTRIDE * 2);                      // [128][ASTRIDE]
  float* Glds = (float*)(smem + 128 * WSTRIDE * 2 + 128 * ASTRIDE * 2);// [128][GSTRIDE]

  const int tid  = threadIdx.x;
  const int gb   = blockIdx.x;
  const int lane = tid & 31;
  const int wv   = tid >> 5;
  const int mg   = wv & 3;   // 4 row groups of 32
  const int ng   = wv >> 2;  // 2 col groups of 64
  const u32 alds_off = (u32)(size_t)(void*)Alds;   // LDS flat addr low 32 = LDS offset

  // Stage this block's 128x768 bf16 weight slice into LDS (stays for all 512 steps)
  for (int i = tid; i < 128 * KCAT; i += blockDim.x) {
    int lr = i / KCAT, k = i % KCAT;
    int q = lr >> 5, j = lr & 31;
    Wlds[lr * WSTRIDE + k] = wcat[(size_t)(q * H_ + gb * 32 + j) * KCAT + k];
  }

  // per-thread cell state: column hc, 16 rows
  const int hc    = tid & 31;
  const int rbase = (tid >> 5) * 16;
  float creg[16];
#pragma unroll
  for (int e = 0; e < 16; ++e)
    creg[e] = cell[(size_t)(rbase + e) * H_ + gb * 32 + hc];

  const float bi = bsum[0 * H_ + gb * 32 + hc];
  const float bf = bsum[1 * H_ + gb * 32 + hc];
  const float bg = bsum[2 * H_ + gb * 32 + hc];
  const float bo = bsum[3 * H_ + gb * 32 + hc];

  __syncthreads();

  for (int t = 0; t < T_; ++t) {
    v8f acc[2][4];
#pragma unroll
    for (int mt = 0; mt < 2; ++mt)
#pragma unroll
      for (int nt = 0; nt < 4; ++nt) acc[mt][nt] = (v8f){};

    const u16* hprev = hbuf + (size_t)(t & 1) * B_ * H_;

    // prefetch next timestep's x into L2 while we grind on this one
    if (t + 1 < T_) {
      const float* pf = state + (size_t)(t + 1) * B_ * D_ + (size_t)(tid >> 1) * D_ +
                        (tid & 1) * 128;
      __builtin_prefetch(pf, 0, 1);
    }

    for (int kc = 0; kc < KCAT / 32; ++kc) {
      __syncthreads();
      if (kc * 32 < D_) {
        // x chunk: f32 -> bf16 conversion, manual staging
        int row = tid >> 1;
        int jb  = (tid & 1) * 16;
        const float* src = state + (size_t)t * B_ * D_ + (size_t)row * D_ + kc * 32 + jb;
#pragma unroll
        for (int e = 0; e < 16; ++e) Alds[row * ASTRIDE + jb + e] = f2bf(src[e]);
      } else {
        // h chunk: 2D TDM tile [128 x 32] bf16, LDS-padded to ASTRIDE pitch
        if (wv == 0) {
          tdm_load_tile32_bf16(hprev + (kc * 32 - D_), alds_off, 128, H_);
          __builtin_amdgcn_s_wait_tensorcnt(0);
        }
      }
      __syncthreads();

      v16bf afr[2], bfr[4];
#pragma unroll
      for (int mt = 0; mt < 2; ++mt) {
        int row = mg * 32 + mt * 16 + (lane & 15);
        afr[mt] = *(const v16bf*)&Alds[row * ASTRIDE + (lane >> 4) * 16];
      }
#pragma unroll
      for (int nt = 0; nt < 4; ++nt) {
        int lr  = ng * 64 + nt * 16 + (lane & 15);
        bfr[nt] = *(const v16bf*)&Wlds[lr * WSTRIDE + kc * 32 + (lane >> 4) * 16];
      }
#pragma unroll
      for (int mt = 0; mt < 2; ++mt)
#pragma unroll
        for (int nt = 0; nt < 4; ++nt)
          acc[mt][nt] = wmma_bf16(afr[mt], bfr[nt], acc[mt][nt]);
    }

    // scatter raw gates to LDS so each thread can combine i/f/g/o
    __syncthreads();
#pragma unroll
    for (int mt = 0; mt < 2; ++mt)
#pragma unroll
      for (int nt = 0; nt < 4; ++nt) {
        int nb = ng * 64 + nt * 16 + (lane & 15);
        int mb = mg * 32 + mt * 16 + (lane >> 4) * 8;
#pragma unroll
        for (int v = 0; v < 8; ++v) Glds[(mb + v) * GSTRIDE + nb] = acc[mt][nt][v];
      }
    __syncthreads();

    u16* hnext = hbuf + (size_t)((t + 1) & 1) * B_ * H_;
#pragma unroll
    for (int e = 0; e < 16; ++e) {
      int row  = rbase + e;
      float gi = Glds[row * GSTRIDE + 0 + hc] + bi;
      float gf = Glds[row * GSTRIDE + 32 + hc] + bf;
      float gg = Glds[row * GSTRIDE + 64 + hc] + bg;
      float go = Glds[row * GSTRIDE + 96 + hc] + bo;
      float cn = sigm(gf) * creg[e] + sigm(gi) * tanhf(gg);
      creg[e]  = cn;
      u16 hb   = f2bf(sigm(go) * tanhf(cn));
      hnext[(size_t)row * H_ + gb * 32 + hc]                    = hb;
      hs[(size_t)t * B_ * H_ + (size_t)row * H_ + gb * 32 + hc] = hb;
    }

    // device-scope grid barrier (16 co-resident blocks).
    // s_cluster_barrier is the native path when dispatched as a cluster and an
    // architectural NOP (ClusterID==0) otherwise; the atomic counter is the
    // portable fallback that always enforces the cross-block ordering.
    __threadfence();
    __syncthreads();
    __builtin_amdgcn_s_cluster_barrier();
    if (tid == 0) {
      __hip_atomic_fetch_add(cnt, 1u, __ATOMIC_ACQ_REL, __HIP_MEMORY_SCOPE_AGENT);
      u32 target = (u32)(t + 1) * (u32)gridDim.x;
      while (__hip_atomic_load(cnt, __ATOMIC_ACQUIRE, __HIP_MEMORY_SCOPE_AGENT) < target)
        __builtin_amdgcn_s_sleep(1);
    }
    __syncthreads();
  }
}

// ---------------------------------------------------------------------------
// Fused 3-layer MLP over 128 packed rows per block; activations stay in LDS.
// ---------------------------------------------------------------------------
__global__ void __launch_bounds__(256)
k_mlp(const float* __restrict__ state, const u16* __restrict__ hs,
      const int* __restrict__ rowmap,
      const u16* __restrict__ w0, const float* __restrict__ b0,
      const u16* __restrict__ w1, const float* __restrict__ b1,
      const u16* __restrict__ w2, const float* __restrict__ b2,
      float* __restrict__ out, int outcols, int do_tanh, int N) {
  extern __shared__ char smem[];
  u16* Y0   = (u16*)smem;                                  // [128][YSTRIDE]
  u16* Y1   = (u16*)(smem + 128 * YSTRIDE * 2);            // [128][YSTRIDE]
  u16* Alds = (u16*)(smem + 2 * 128 * YSTRIDE * 2);        // [128][ASTRIDE]
  int* rm   = (int*)(smem + 2 * 128 * YSTRIDE * 2 + 128 * ASTRIDE * 2);

  const int tid  = threadIdx.x;
  const int lane = tid & 31;
  const int wv   = tid >> 5;
  const int mg   = wv & 3;
  const int ng   = wv >> 2;
  const int rowbase = blockIdx.x * 128;

  if (tid < 128) {
    int n   = rowbase + tid;
    rm[tid] = (n < N) ? rowmap[n] : 0;
  }
  __syncthreads();

  // ---- layer 0: gathered [x | h] (K=768) -> 512, ReLU -> Y0 ----
  for (int np = 0; np < 4; ++np) {
    v8f acc[2][4];
#pragma unroll
    for (int mt = 0; mt < 2; ++mt)
#pragma unroll
      for (int nt = 0; nt < 4; ++nt) acc[mt][nt] = (v8f){};

    for (int kc = 0; kc < KCAT / 32; ++kc) {
      __syncthreads();
      {
        int row = tid >> 1;
        int jb  = (tid & 1) * 16;
        int kk0 = kc * 32 + jb;
        int v   = rm[row];
        int bt  = v >> 9;    // batch
        int tt  = v & 511;   // time
        if (kk0 < D_) {
          const float* src = state + (size_t)tt * B_ * D_ + (size_t)bt * D_ + kk0;
#pragma unroll
          for (int e = 0; e < 16; ++e) Alds[row * ASTRIDE + jb + e] = f2bf(src[e]);
        } else {
          const u16* src = hs + (size_t)tt * B_ * H_ + (size_t)bt * H_ + (kk0 - D_);
#pragma unroll
          for (int e = 0; e < 16; ++e) Alds[row * ASTRIDE + jb + e] = src[e];
        }
      }
      __syncthreads();

      v16bf afr[2], bfr[4];
#pragma unroll
      for (int mt = 0; mt < 2; ++mt) {
        int row = mg * 32 + mt * 16 + (lane & 15);
        afr[mt] = *(const v16bf*)&Alds[row * ASTRIDE + (lane >> 4) * 16];
      }
#pragma unroll
      for (int nt = 0; nt < 4; ++nt) {
        int nc  = np * 128 + ng * 64 + nt * 16 + (lane & 15);
        bfr[nt] = *(const v16bf*)&w0[(size_t)nc * KCAT + kc * 32 + (lane >> 4) * 16];
      }
#pragma unroll
      for (int mt = 0; mt < 2; ++mt)
#pragma unroll
        for (int nt = 0; nt < 4; ++nt)
          acc[mt][nt] = wmma_bf16(afr[mt], bfr[nt], acc[mt][nt]);
    }
#pragma unroll
    for (int mt = 0; mt < 2; ++mt)
#pragma unroll
      for (int nt = 0; nt < 4; ++nt) {
        int nc   = np * 128 + ng * 64 + nt * 16 + (lane & 15);
        int mb   = mg * 32 + mt * 16 + (lane >> 4) * 8;
        float bb = b0[nc];
#pragma unroll
        for (int v = 0; v < 8; ++v) {
          float val = fmaxf(acc[mt][nt][v] + bb, 0.0f);
          Y0[(mb + v) * YSTRIDE + nc] = f2bf(val);
        }
      }
  }
  __syncthreads();

  // ---- layer 1: Y0 (K=512) -> 512, ReLU -> Y1 ----
  for (int np = 0; np < 4; ++np) {
    v8f acc[2][4];
#pragma unroll
    for (int mt = 0; mt < 2; ++mt)
#pragma unroll
      for (int nt = 0; nt < 4; ++nt) acc[mt][nt] = (v8f){};

    for (int kc = 0; kc < MID_ / 32; ++kc) {
      v16bf afr[2], bfr[4];
#pragma unroll
      for (int mt = 0; mt < 2; ++mt) {
        int row = mg * 32 + mt * 16 + (lane & 15);
        afr[mt] = *(const v16bf*)&Y0[row * YSTRIDE + kc * 32 + (lane >> 4) * 16];
      }
#pragma unroll
      for (int nt = 0; nt < 4; ++nt) {
        int nc  = np * 128 + ng * 64 + nt * 16 + (lane & 15);
        bfr[nt] = *(const v16bf*)&w1[(size_t)nc * MID_ + kc * 32 + (lane >> 4) * 16];
      }
#pragma unroll
      for (int mt = 0; mt < 2; ++mt)
#pragma unroll
        for (int nt = 0; nt < 4; ++nt)
          acc[mt][nt] = wmma_bf16(afr[mt], bfr[nt], acc[mt][nt]);
    }
#pragma unroll
    for (int mt = 0; mt < 2; ++mt)
#pragma unroll
      for (int nt = 0; nt < 4; ++nt) {
        int nc   = np * 128 + ng * 64 + nt * 16 + (lane & 15);
        int mb   = mg * 32 + mt * 16 + (lane >> 4) * 8;
        float bb = b1[nc];
#pragma unroll
        for (int v = 0; v < 8; ++v) {
          float val = fmaxf(acc[mt][nt][v] + bb, 0.0f);
          Y1[(mb + v) * YSTRIDE + nc] = f2bf(val);
        }
      }
  }
  __syncthreads();

  // ---- layer 2: Y1 (K=512) -> 16 (zero-padded head); one M-tile per wave ----
  {
    v8f acc2 = (v8f){};
    for (int kc = 0; kc < MID_ / 32; ++kc) {
      int row  = wv * 16 + (lane & 15);
      v16bf a  = *(const v16bf*)&Y1[row * YSTRIDE + kc * 32 + (lane >> 4) * 16];
      v16bf b  = *(const v16bf*)&w2[(size_t)(lane & 15) * MID_ + kc * 32 + (lane >> 4) * 16];
      acc2 = wmma_bf16(a, b, acc2);
    }
    int n = lane & 15;
    if (n < outcols) {
      float bb = b2[n];
#pragma unroll
      for (int v = 0; v < 8; ++v) {
        int m  = wv * 16 + (lane >> 4) * 8 + v;
        int gr = rowbase + m;
        if (gr < N) {
          float val = acc2[v] + bb;
          if (do_tanh) val = tanhf(val);
          out[(size_t)gr * outcols + n] = val;
        }
      }
    }
  }
}

// ---------------------------------------------------------------------------
extern "C" void kernel_launch(void* const* d_in, const int* in_sizes, int n_in,
                              void* d_out, int out_size, void* d_ws, size_t ws_size,
                              hipStream_t stream) {
  (void)in_sizes; (void)n_in; (void)ws_size;

  const float* state  = (const float*)d_in[0];
  const float* hidden = (const float*)d_in[1];
  const float* cell   = (const float*)d_in[2];
  const int*   lens   = (const int*)d_in[3];
  const float* W_ih   = (const float*)d_in[4];
  const float* W_hh   = (const float*)d_in[5];
  const float* b_ih   = (const float*)d_in[6];
  const float* b_hh   = (const float*)d_in[7];
  const float* aw0 = (const float*)d_in[8];  const float* ab0 = (const float*)d_in[9];
  const float* aw1 = (const float*)d_in[10]; const float* ab1 = (const float*)d_in[11];
  const float* aw2 = (const float*)d_in[12]; const float* ab2 = (const float*)d_in[13];
  const float* cw0 = (const float*)d_in[14]; const float* cb0 = (const float*)d_in[15];
  const float* cw1 = (const float*)d_in[16]; const float* cb1 = (const float*)d_in[17];
  const float* cw2 = (const float*)d_in[18]; const float* cb2 = (const float*)d_in[19];

  char* ws = (char*)d_ws;
  size_t off = 0;
  auto take = [&](size_t bytes) -> char* {
    char* p = ws + off;
    off += (bytes + 255) & ~(size_t)255;
    return p;
  };
  u16*   wcat   = (u16*)take((size_t)G4H * KCAT * 2);
  float* bsum   = (float*)take((size_t)G4H * 4);
  u16*   hbuf   = (u16*)take((size_t)2 * B_ * H_ * 2);
  u16*   hs     = (u16*)take((size_t)T_ * B_ * H_ * 2);
  int*   offs   = (int*)take((size_t)(B_ + 1) * 4);
  int*   rowmap = (int*)take((size_t)B_ * T_ * 4);
  u16*   w0a    = (u16*)take((size_t)MID_ * KCAT * 2);
  u16*   w1a    = (u16*)take((size_t)MID_ * MID_ * 2);
  u16*   w2a    = (u16*)take((size_t)16 * MID_ * 2);
  u16*   w0c    = (u16*)take((size_t)MID_ * KCAT * 2);
  u16*   w1c    = (u16*)take((size_t)MID_ * MID_ * 2);
  u16*   w2c    = (u16*)take((size_t)16 * MID_ * 2);
  u32*   cnt    = (u32*)take(4);

  const int N = out_size / 4;   // actor [N,3] + critic [N,1]

  // --- prep ---
  k_prep_lstm<<<1024, 256, 0, stream>>>(W_ih, W_hh, b_ih, b_hh, hidden,
                                        wcat, bsum, hbuf, cnt);
  k_cvt<<<512, 256, 0, stream>>>(w0a, aw0, MID_ * KCAT);
  k_cvt<<<512, 256, 0, stream>>>(w1a, aw1, MID_ * MID_);
  k_cvt<<<512, 256, 0, stream>>>(w0c, cw0, MID_ * KCAT);
  k_cvt<<<512, 256, 0, stream>>>(w1c, cw1, MID_ * MID_);
  k_pad16<<<32, 256, 0, stream>>>(w2a, aw2, 3);
  k_pad16<<<32, 256, 0, stream>>>(w2c, cw2, 1);
  k_offsets<<<1, 32, 0, stream>>>(lens, offs);
  k_rowmap<<<B_, 256, 0, stream>>>(lens, offs, rowmap);

  // --- persistent LSTM ---
  size_t lstm_lds = (size_t)128 * WSTRIDE * 2 + (size_t)128 * ASTRIDE * 2 +
                    (size_t)128 * GSTRIDE * 4;
  (void)hipFuncSetAttribute((const void*)k_lstm,
                            hipFuncAttributeMaxDynamicSharedMemorySize, (int)lstm_lds);
  k_lstm<<<LSTM_BLOCKS, 256, lstm_lds, stream>>>(state, cell, wcat, bsum, hbuf, hs, cnt);

  // --- fused actor / critic MLPs ---
  size_t mlp_lds = (size_t)2 * 128 * YSTRIDE * 2 + (size_t)128 * ASTRIDE * 2 +
                   (size_t)128 * 4;
  (void)hipFuncSetAttribute((const void*)k_mlp,
                            hipFuncAttributeMaxDynamicSharedMemorySize, (int)mlp_lds);
  int grid = (N + 127) / 128;
  if (grid < 1) grid = 1;
  float* actor_out  = (float*)d_out;
  float* critic_out = (float*)d_out + (size_t)3 * N;
  k_mlp<<<grid, 256, mlp_lds, stream>>>(state, hs, rowmap, w0a, ab0, w1a, ab1,
                                        w2a, ab2, actor_out, 3, 1, N);
  k_mlp<<<grid, 256, mlp_lds, stream>>>(state, hs, rowmap, w0c, cb0, w1c, cb1,
                                        w2c, cb2, critic_out, 1, 0, N);
}
